// ArabicStructuralPositionEncoder_81724637708484
// MI455X (gfx1250) — compile-verified, matrix operand order checked by reading, not run on hardware
//
#include <hip/hip_runtime.h>
#include <hip/hip_bf16.h>
#include <hip/hip_fp16.h>

typedef __attribute__((ext_vector_type(16))) _Float16 v16h;
typedef __attribute__((ext_vector_type(8)))  float    v8f;
typedef unsigned int u32;

#define Bn   32
#define Wn   2048
#define Dn   1024
#define DQn  256
#define BIGI 1000000

// ------------------------------------------------------------------
// Kernel 1: per-batch-row scans -> per-token feature indices.
// One block (256 threads) per batch row; LDS Hillis-Steele scans.
// feat[token] = { depth_idx, vdist_idx, conj_idx, bitcast(rel) }
// ------------------------------------------------------------------
__global__ void k_scan(const int* __restrict__ pos_tags,
                       const int* __restrict__ seq_lengths,
                       int4* __restrict__ feat) {
  __shared__ int s[256];
  __shared__ int s_left[Wn];
  const int b = blockIdx.x;
  const int t = threadIdx.x;
  const int* row = pos_tags + b * Wn;
  const float invLen = 1.0f / fmaxf((float)seq_lengths[b], 1.0f);

  int cd = 0, cc = 0, cl = -BIGI;      // carries (uniform across block)
  for (int c = 0; c < Wn / 256; ++c) {
    const int w   = c * 256 + t;
    const int tag = row[w];
    const int vd  = (tag == 15) ? 1 : 0;               // SCONJ
    const int vc  = (tag == 9)  ? 1 : 0;               // CC
    const bool isv = (tag == 10) || (tag == 11);       // VERB_A / VERB_B

    // inclusive add-scan: depth
    __syncthreads(); s[t] = vd; __syncthreads();
    for (int o = 1; o < 256; o <<= 1) { int x = (t >= o) ? s[t - o] : 0; __syncthreads(); s[t] += x; __syncthreads(); }
    const int depth = s[t] + cd; const int cd2 = s[255] + cd;

    // inclusive add-scan: conj
    __syncthreads(); s[t] = vc; __syncthreads();
    for (int o = 1; o < 256; o <<= 1) { int x = (t >= o) ? s[t - o] : 0; __syncthreads(); s[t] += x; __syncthreads(); }
    const int conj = s[t] + cc; const int cc2 = s[255] + cc;

    // inclusive max-scan: nearest verb to the left
    __syncthreads(); s[t] = isv ? w : -BIGI; __syncthreads();
    for (int o = 1; o < 256; o <<= 1) { int x = (t >= o) ? s[t - o] : -BIGI; __syncthreads(); s[t] = max(s[t], x); __syncthreads(); }
    const int left = max(s[t], cl); const int cl2 = max(s[255], cl);

    cd = cd2; cc = cc2; cl = cl2;
    s_left[w] = left;

    int4 f;
    f.x = min(depth, 7);
    f.z = min(conj, 7);
    f.w = __float_as_int((float)w * invLen);
    f.y = 0;
    feat[b * Wn + w] = f;
  }
  const bool any_verb = (cl > -BIGI);

  // backward pass: nearest verb to the right (reverse cummin), then combine
  int cr = BIGI;
  for (int c = Wn / 256 - 1; c >= 0; --c) {
    const int w = c * 256 + (255 - t);   // t increasing == w decreasing
    const int tag = row[w];
    const bool isv = (tag == 10) || (tag == 11);

    __syncthreads(); s[t] = isv ? w : BIGI; __syncthreads();
    for (int o = 1; o < 256; o <<= 1) { int x = (t >= o) ? s[t - o] : BIGI; __syncthreads(); s[t] = min(s[t], x); __syncthreads(); }
    const int right = min(s[t], cr); cr = min(s[255], cr);

    const int left = s_left[w];
    const int dl = w - left;     // >= 0 (huge if none)
    const int dr = w - right;    // <= 0 (huge-negative if none)
    const int adl = dl < 0 ? -dl : dl;
    const int adr = dr < 0 ? -dr : dr;
    int v = (adl <= adr) ? dl : dr;
    if (!any_verb) v = 0;
    v = min(max(v, -16), 16) + 16;       // -> [0, 32]
    feat[b * Wn + w].y = v;
  }
}

// ------------------------------------------------------------------
// Kernel 2: fus_W f32 -> f16 (packed pairs), same [N][K] row layout
// ------------------------------------------------------------------
__global__ void k_wconv(const float* __restrict__ w, u32* __restrict__ wh) {
  const int p = blockIdx.x * blockDim.x + threadIdx.x;   // pair index
  union { _Float16 h[2]; u32 u; } pk;
  pk.h[0] = (_Float16)w[2 * p];
  pk.h[1] = (_Float16)w[2 * p + 1];
  wh[p] = pk.u;
}

// ------------------------------------------------------------------
// Kernel 3: gather embeddings -> x f16 [B*W][D]; 2 halves per thread
// ------------------------------------------------------------------
__global__ void k_build_x(const int4* __restrict__ feat,
                          const float* __restrict__ depth_emb,
                          const float* __restrict__ vdist_emb,
                          const float* __restrict__ conj_emb,
                          const float* __restrict__ relW,
                          const float* __restrict__ relB,
                          u32* __restrict__ xh) {
  const int gid   = blockIdx.x * blockDim.x + threadIdx.x;   // pair index
  const int token = gid >> 9;             // 512 pairs per token
  const int d     = (gid & 511) << 1;
  const int4 f = feat[token];
  const int seg = d >> 8, off = d & 255;
  float a, b;
  if      (seg == 0) { const float* e = depth_emb + f.x * DQn + off; a = e[0]; b = e[1]; }
  else if (seg == 1) { const float* e = vdist_emb + f.y * DQn + off; a = e[0]; b = e[1]; }
  else if (seg == 2) { const float* e = conj_emb  + f.z * DQn + off; a = e[0]; b = e[1]; }
  else {
    const float r = __int_as_float(f.w);
    a = r * relW[off]     + relB[off];
    b = r * relW[off + 1] + relB[off + 1];
  }
  union { _Float16 h[2]; u32 u; } pk;
  pk.h[0] = (_Float16)a; pk.h[1] = (_Float16)b;
  xh[gid] = pk.u;
}

// ------------------------------------------------------------------
// Kernel 4: fused WMMA GEMM + bias + gelu + LayerNorm.
// One block per M-tile (16 tokens); block covers the FULL N=1024, so
// LayerNorm statistics are computed in-block and h never hits HBM.
// 8 waves x 8 N-tiles each; each wave reuses one A-frag for 8 WMMAs
// per k-step (8x better load/FLOP ratio; A amplification 64x -> 8x,
// all L2-resident).
// C/D layout per ISA: VGPR r, lane L -> row (r + 8*(L>>4)), col (L&15).
// ------------------------------------------------------------------
__global__ void k_gemm_ln(const _Float16* __restrict__ X,    // [65536][1024]
                          const _Float16* __restrict__ Wh,   // [1024][1024]
                          const float* __restrict__ bias,
                          const float* __restrict__ lng,
                          const float* __restrict__ lnb,
                          float* __restrict__ out) {         // [65536][1024]
  __shared__ float rs[16], rq[16];       // per-row sum / sumsq
  __shared__ float smu[16], sinv[16];

  const int wave = threadIdx.x >> 5;     // 0..7
  const int lane = threadIdx.x & 31;
  const int mtile = blockIdx.x;          // 4096 blocks
  const int m  = lane & 15;
  const int hi = lane >> 4;

  if (threadIdx.x < 16) { rs[threadIdx.x] = 0.0f; rq[threadIdx.x] = 0.0f; }
  __syncthreads();

  const u32* arow  = (const u32*)(X  + (size_t)(mtile * 16 + m) * Dn);
  const u32* brow0 = (const u32*)(Wh + (size_t)((wave * 8) * 16 + m) * Dn);
  const int  bstep = 16 * Dn / 2;        // dwords between consecutive n-tiles

  v8f acc[8];
  #pragma unroll
  for (int j = 0; j < 8; ++j) acc[j] = (v8f){};

  union FragU { v16h h; uint4 q[2]; };

  for (int k0 = 0; k0 < Dn; k0 += 32) {
    const int base = (k0 >> 1) + hi * 4;   // dword index within a row
    FragU fa;
    fa.q[0] = *(const uint4*)(arow + base);
    fa.q[1] = *(const uint4*)(arow + base + 8);
    #pragma unroll
    for (int j = 0; j < 8; ++j) {
      const u32* br = brow0 + j * bstep;
      FragU fb;
      fb.q[0] = *(const uint4*)(br + base);
      fb.q[1] = *(const uint4*)(br + base + 8);
      acc[j] = __builtin_amdgcn_wmma_f32_16x16x32_f16(
          false, fa.h, false, fb.h, (short)0, acc[j], false, false);
    }
  }

  // bias + exact gelu, accumulate per-row partial stats
  float ps[8], pq[8];
  #pragma unroll
  for (int r = 0; r < 8; ++r) { ps[r] = 0.0f; pq[r] = 0.0f; }
  #pragma unroll
  for (int j = 0; j < 8; ++j) {
    const int col = (wave * 8 + j) * 16 + m;
    const float bv = bias[col];
    #pragma unroll
    for (int r = 0; r < 8; ++r) {
      const float v = acc[j][r] + bv;
      const float ge = 0.5f * v * (1.0f + erff(v * 0.70710678118f));
      acc[j][r] = ge;
      ps[r] += ge;
      pq[r] += ge * ge;
    }
  }
  #pragma unroll
  for (int r = 0; r < 8; ++r) {
    atomicAdd(&rs[hi * 8 + r], ps[r]);   // -> ds_add_f32
    atomicAdd(&rq[hi * 8 + r], pq[r]);
  }
  __syncthreads();

  if (threadIdx.x < 16) {
    const float mu  = rs[threadIdx.x] * (1.0f / Dn);
    const float var = rq[threadIdx.x] * (1.0f / Dn) - mu * mu;
    smu[threadIdx.x]  = mu;
    sinv[threadIdx.x] = rsqrtf(var + 1e-5f);
  }
  __syncthreads();

  // normalize in registers, single store of the final output
  #pragma unroll
  for (int j = 0; j < 8; ++j) {
    const int col = (wave * 8 + j) * 16 + m;
    const float gg = lng[col];
    const float bb = lnb[col];
    #pragma unroll
    for (int r = 0; r < 8; ++r) {
      const int row = hi * 8 + r;
      out[(size_t)(mtile * 16 + row) * Dn + col] =
          (acc[j][r] - smu[row]) * sinv[row] * gg + bb;
    }
  }
}

// ------------------------------------------------------------------
extern "C" void kernel_launch(void* const* d_in, const int* in_sizes, int n_in,
                              void* d_out, int out_size, void* d_ws, size_t ws_size,
                              hipStream_t stream) {
  const int*   pos_tags  = (const int*)d_in[1];
  const int*   seql      = (const int*)d_in[2];
  const float* depth_emb = (const float*)d_in[4];
  const float* vdist_emb = (const float*)d_in[5];
  const float* conj_emb  = (const float*)d_in[6];
  const float* relW      = (const float*)d_in[7];
  const float* relB      = (const float*)d_in[8];
  const float* fusW      = (const float*)d_in[9];
  const float* fusB      = (const float*)d_in[10];
  const float* lng       = (const float*)d_in[11];
  const float* lnb       = (const float*)d_in[12];

  char* ws = (char*)d_ws;
  int4*     feat = (int4*)ws;                            // 1 MB
  _Float16* Wh   = (_Float16*)(ws + (1u << 20));         // 2 MB
  _Float16* X    = (_Float16*)(ws + (3u << 20));         // 128 MB
  float*    out  = (float*)d_out;

  k_scan   <<<Bn, 256, 0, stream>>>(pos_tags, seql, feat);
  k_wconv  <<<(Dn * Dn / 2) / 256, 256, 0, stream>>>(fusW, (u32*)Wh);
  k_build_x<<<((size_t)Bn * Wn * Dn / 2) / 256, 256, 0, stream>>>(
      feat, depth_emb, vdist_emb, conj_emb, relW, relB, (u32*)X);
  k_gemm_ln<<<Bn * Wn / 16, 256, 0, stream>>>(X, Wh, fusB, lng, lnb, out);
}
